// _GraphNonLocal_29867202576703
// MI455X (gfx1250) — matrix operand order repaired and checked by reading, not compiled, not run on hardware
//
#include <hip/hip_runtime.h>

typedef __attribute__((ext_vector_type(16))) __bf16 v16bf;
typedef __attribute__((ext_vector_type(8)))  __bf16 v8bf;
typedef __attribute__((ext_vector_type(8)))  float  v8f;

namespace {

constexpr int kN  = 64;    // tokens
constexpr int kC  = 256;   // channels
constexpr int kI  = 128;   // inner dim
constexpr int kM  = 16;    // pooled tokens (N/4)
constexpr int LDX = 260;   // f32 pitch of staged x   (260 % 64 == 4 -> conflict free)
constexpr int LDT = 132;   // f32 pitch of theta
constexpr int LDP = 132;   // f32 pitch of pooled phi/g
constexpr int LDY = 136;   // bf16 pitch of y^T       (136h = 272B -> 68 dw % 64 == 4)

__device__ __forceinline__ v8f wmma_bf16(v16bf a, v16bf b, v8f c) {
  // D = A(16x32) * B(32x16) + C, fp32 accum
  return __builtin_amdgcn_wmma_f32_16x16x32_bf16(false, a, false, b, (short)0, c,
                                                 false, false);
}

// A fragment (16x32, bf16) built from a row-major fp32 LDS tile.
// Contiguous K order used here (B matches): lane<16 holds K {k0..k0+15}; lane>=16 holds K {k0+16..k0+31}.
__device__ __forceinline__ v16bf a_frag_f32(const float* row, int k0, int hi) {
  const float* r = row + k0 + hi * 16;
  const float4 f0 = *(const float4*)(r + 0);
  const float4 f1 = *(const float4*)(r + 4);
  const float4 f2 = *(const float4*)(r + 8);
  const float4 f3 = *(const float4*)(r + 12);
  v16bf a;
  a[0]  = (__bf16)f0.x; a[1]  = (__bf16)f0.y; a[2]  = (__bf16)f0.z; a[3]  = (__bf16)f0.w;
  a[4]  = (__bf16)f1.x; a[5]  = (__bf16)f1.y; a[6]  = (__bf16)f1.z; a[7]  = (__bf16)f1.w;
  a[8]  = (__bf16)f2.x; a[9]  = (__bf16)f2.y; a[10] = (__bf16)f2.z; a[11] = (__bf16)f2.w;
  a[12] = (__bf16)f3.x; a[13] = (__bf16)f3.y; a[14] = (__bf16)f3.z; a[15] = (__bf16)f3.w;
  return a;
}

// A fragment from a row-major bf16 LDS tile.
__device__ __forceinline__ v16bf a_frag_bf16(const __bf16* row, int k0, int hi) {
  const __bf16* r = row + k0 + hi * 16;
  const v8bf lo  = *(const v8bf*)(r);
  const v8bf hi8 = *(const v8bf*)(r + 8);
  v16bf a;
#pragma unroll
  for (int j = 0; j < 8; ++j) { a[j] = lo[j]; a[8 + j] = hi8[j]; }
  return a;
}

}  // namespace

// One-shot: convert the four weight matrices to bf16 in workspace.
// Layout: [0)=theta_w, [32768)=phi_w, [65536)=g_w, [98304)=w_w  (131072 bf16)
__global__ void prep_weights_kernel(const float* __restrict__ thw,
                                    const float* __restrict__ phw,
                                    const float* __restrict__ gw,
                                    const float* __restrict__ ww,
                                    __bf16* __restrict__ dst) {
  const int i = blockIdx.x * blockDim.x + threadIdx.x;  // 0 .. 131071
  float v;
  if (i < 32768)      v = thw[i];
  else if (i < 65536) v = phw[i - 32768];
  else if (i < 98304) v = gw[i - 65536];
  else                v = ww[i - 98304];
  dst[i] = (__bf16)v;
}

__global__ __launch_bounds__(256) void fused_nonlocal_kernel(
    const float* __restrict__ x,
    const int*   __restrict__ grouped,
    const float* __restrict__ theta_b,
    const float* __restrict__ phi_b,
    const float* __restrict__ g_b,
    const float* __restrict__ w_b,
    const float* __restrict__ bn_gamma,
    const float* __restrict__ bn_beta,
    const __bf16* __restrict__ Wq,   // [3][128*256] theta/phi/g, row-major (I,C)
    const __bf16* __restrict__ Ww,   // [256*128]    w_w, row-major (C,I)
    float* __restrict__ out) {
  __shared__ float  xg[kN * LDX];      // 66560 B (fp32 residual + GEMM1 A source)
  __shared__ float  theta[kN * LDT];   // 33792 B
  __shared__ float  phiP[kM * LDP];    //  8448 B
  __shared__ float  gxP[kM * LDP];     //  8448 B
  __shared__ float  attn[kN * 16];     //  4096 B
  __shared__ __bf16 yT[kN * LDY];      // 17408 B
  __shared__ int    gord[kN];

  const int b    = blockIdx.x;
  const int t    = threadIdx.x;
  const int wave = t >> 5;
  const int lane = t & 31;
  const int lm   = lane & 15;
  const int hi   = lane >> 4;

  if (t < kN) gord[t] = grouped[t];
  __syncthreads();

  // ---- stage gathered x rows into LDS via async global->LDS DMA -------
  // (CDNA5 path: no VGPR round trip, tracked by ASYNCcnt)
  const float* xb = x + (size_t)b * (kN * kC);
  for (int j = t; j < kN * kC / 4; j += 256) {
    const int n  = j >> 6;
    const int c4 = j & 63;
    const float*   src  = xb + (size_t)gord[n] * kC + c4 * 4;
    const uint32_t ldsa = (uint32_t)(uintptr_t)(&xg[n * LDX + c4 * 4]);
    asm volatile("global_load_async_to_lds_b128 %0, %1, off"
                 :: "v"(ldsa), "v"(src) : "memory");
  }
  asm volatile("s_wait_asynccnt 0x0" ::: "memory");
  __syncthreads();

  // ---- GEMM1 x3: (64x256)@(256x128) for theta / phi / g ---------------
  // wave w owns I-strip [16w, 16w+16); 4 M-tiles; K loop of 8 x 32.
  // B fragments are software-pipelined one k-step ahead so the global-load
  // latency is covered by 4 WMMAs + A-fragment cvt work.
  const int i0 = wave * 16;
#pragma unroll 1
  for (int g = 0; g < 3; ++g) {
    const __bf16* W = Wq + g * (kI * kC) + (size_t)(i0 + lm) * kC + hi * 16;
    v8f acc[4] = {};
    v16bf bcur = *(const v16bf*)(W);  // B col = weight row, contiguous K
#pragma unroll
    for (int kk = 0; kk < 8; ++kk) {
      const int k0 = kk * 32;
      const v16bf bnext = (kk < 7) ? *(const v16bf*)(W + k0 + 32) : bcur;
#pragma unroll
      for (int mt = 0; mt < 4; ++mt) {
        const v16bf af = a_frag_f32(&xg[(mt * 16 + lm) * LDX], k0, hi);
        acc[mt] = wmma_bf16(af, bcur, acc[mt]);
      }
      bcur = bnext;
    }
    if (g == 0) {
      const float bias = theta_b[i0 + lm];
#pragma unroll
      for (int mt = 0; mt < 4; ++mt)
#pragma unroll
        for (int vg = 0; vg < 8; ++vg)
          theta[(mt * 16 + vg + hi * 8) * LDT + i0 + lm] = acc[mt][vg] + bias;
    } else {
      // max-pool groups of 4 rows: rows {0-3},{4-7} in this half-wave's VGPRs
      const float bias = (g == 1) ? phi_b[i0 + lm] : g_b[i0 + lm];
      float* dst = (g == 1) ? phiP : gxP;
#pragma unroll
      for (int mt = 0; mt < 4; ++mt) {
        const float pa = fmaxf(fmaxf(acc[mt][0], acc[mt][1]), fmaxf(acc[mt][2], acc[mt][3]));
        const float pb = fmaxf(fmaxf(acc[mt][4], acc[mt][5]), fmaxf(acc[mt][6], acc[mt][7]));
        const int mrow = mt * 4 + hi * 2;
        dst[mrow * LDP + i0 + lm]       = pa + bias;
        dst[(mrow + 1) * LDP + i0 + lm] = pb + bias;
      }
    }
  }
  __syncthreads();

  // ---- f = theta @ phiP^T (64x16) + softmax over m (fp32 VALU) --------
#pragma unroll 1
  for (int p = 0; p < 4; ++p) {
    const int n = (t >> 4) + 16 * p;  // 16 consecutive lanes share a row
    const int m = t & 15;
    const float* th = &theta[n * LDT];
    const float* ph = &phiP[m * LDP];
    float f = 0.f;
#pragma unroll 4
    for (int i = 0; i < kI; i += 4) {
      const float4 a4 = *(const float4*)(th + i);
      const float4 p4 = *(const float4*)(ph + i);
      f += a4.x * p4.x + a4.y * p4.y + a4.z * p4.z + a4.w * p4.w;
    }
    float mx = f;
#pragma unroll
    for (int d = 8; d >= 1; d >>= 1) mx = fmaxf(mx, __shfl_xor(mx, d, 16));
    const float e = __expf(f - mx);
    float s = e;
#pragma unroll
    for (int d = 8; d >= 1; d >>= 1) s += __shfl_xor(s, d, 16);
    attn[n * 16 + m] = e / s;
  }
  __syncthreads();

  // ---- y^T[n][i] = sum_m a[n][m] * gxP[m][i]  (store bf16) ------------
  for (int j = t; j < kN * kI; j += 256) {
    const int n = j >> 7;
    const int i = j & 127;
    const float* ar = &attn[n * 16];
    float acc = 0.f;
#pragma unroll
    for (int m = 0; m < 16; ++m) acc += ar[m] * gxP[m * LDP + i];
    yT[n * LDY + i] = (__bf16)acc;
  }
  __syncthreads();

  // ---- out_nc = y^T(64x128) @ Ww^T(128x256) + BN affine + residual ----
  const float bnscale = rsqrtf(1.0f + 1e-5f);
  float* outb = out + (size_t)b * (kN * kC);
#pragma unroll 1
  for (int half = 0; half < 2; ++half) {
    const int nt = wave + half * 8;        // C-tile index (16 tiles total)
    const int c  = nt * 16 + lm;
    const __bf16* Wc = Ww + (size_t)c * kI + hi * 16;
    v8f acc[4] = {};
    v16bf bcur = *(const v16bf*)(Wc);
#pragma unroll
    for (int kk = 0; kk < 4; ++kk) {
      const int k0 = kk * 32;
      const v16bf bnext = (kk < 3) ? *(const v16bf*)(Wc + k0 + 32) : bcur;
#pragma unroll
      for (int mt = 0; mt < 4; ++mt) {
        const v16bf af = a_frag_bf16(&yT[(mt * 16 + lm) * LDY], k0, hi);
        acc[mt] = wmma_bf16(af, bcur, acc[mt]);
      }
      bcur = bnext;
    }
    const float sc  = bn_gamma[c] * bnscale;
    const float off = w_b[c] * sc + bn_beta[c];
#pragma unroll
    for (int mt = 0; mt < 4; ++mt) {
#pragma unroll
      for (int vg = 0; vg < 8; ++vg) {
        const int n = mt * 16 + vg + hi * 8;
        const float v = acc[mt][vg] * sc + off + xg[n * LDX + c];
        // restored = grouped^{-1}: grouped row n maps back to original row gord[n]
        outb[(size_t)gord[n] * kC + c] = v;
      }
    }
  }
}

extern "C" void kernel_launch(void* const* d_in, const int* in_sizes, int n_in,
                              void* d_out, int out_size, void* d_ws, size_t ws_size,
                              hipStream_t stream) {
  (void)in_sizes; (void)n_in; (void)out_size; (void)ws_size;
  const float* x       = (const float*)d_in[0];
  const int*   grouped = (const int*)d_in[1];
  // d_in[2] = restored_order (unused: restored = grouped^{-1})
  const float* thw = (const float*)d_in[3];
  const float* thb = (const float*)d_in[4];
  const float* phw = (const float*)d_in[5];
  const float* phb = (const float*)d_in[6];
  const float* gw  = (const float*)d_in[7];
  const float* gb  = (const float*)d_in[8];
  const float* ww  = (const float*)d_in[9];
  const float* wb  = (const float*)d_in[10];
  const float* bng = (const float*)d_in[11];
  const float* bnb = (const float*)d_in[12];

  __bf16* wbf = (__bf16*)d_ws;  // 131072 bf16 = 256 KB
  prep_weights_kernel<<<dim3(512), dim3(256), 0, stream>>>(thw, phw, gw, ww, wbf);
  fused_nonlocal_kernel<<<dim3(4096), dim3(256), 0, stream>>>(
      x, grouped, thb, phb, gb, wb, bng, bnb,
      wbf, wbf + 98304, (float*)d_out);
}
